// SpatialAttention_69329362092714
// MI455X (gfx1250) — matrix operand order repaired and checked by reading, not compiled, not run on hardware
//
#include <hip/hip_runtime.h>
#include <hip/hip_bf16.h>

typedef __attribute__((ext_vector_type(16))) __bf16 v16bf;
typedef __attribute__((ext_vector_type(8)))  __bf16 v8bf;
typedef __attribute__((ext_vector_type(8)))  float  v8f;

#define B_   32
#define IDF_ 512
#define CDF_ 512
#define Q_   4096
#define L_   256

// ---------------------------------------------------------------------------
// flat fp32 -> bf16 convert
// ---------------------------------------------------------------------------
__global__ void cvt_f32_bf16(const float* __restrict__ in, __bf16* __restrict__ out, size_t n) {
    size_t i = (size_t)blockIdx.x * blockDim.x + threadIdx.x;
    size_t stride = (size_t)gridDim.x * blockDim.x;
    for (; i < n; i += stride) out[i] = (__bf16)in[i];
}

// ---------------------------------------------------------------------------
// batched [rows,cols] fp32 -> [cols,rows] bf16 transpose via LDS tile
// ---------------------------------------------------------------------------
__global__ void transpose_f32_to_bf16(const float* __restrict__ in, __bf16* __restrict__ out,
                                      int rows, int cols) {
    __shared__ float tile[32][33];
    size_t b = blockIdx.z;
    const float* ip = in + b * (size_t)rows * cols;
    __bf16* op = out + b * (size_t)rows * cols;
    int c0 = blockIdx.x * 32, r0 = blockIdx.y * 32;
    for (int i = threadIdx.y; i < 32; i += 8)
        tile[i][threadIdx.x] = ip[(size_t)(r0 + i) * cols + c0 + threadIdx.x];
    __syncthreads();
    for (int i = threadIdx.y; i < 32; i += 8)
        op[(size_t)(c0 + i) * rows + r0 + threadIdx.x] = (__bf16)tile[threadIdx.x][i];
}

// ---------------------------------------------------------------------------
// batched bf16 GEMM: D[M,N] = A[M,K] (row-major) x BT[N,K]^T (row-major),
// fp32 accumulate via v_wmma_f32_16x16x32_bf16.  lda = ldbt = K.
// Block: 256 threads = 8 waves; block tile 128(M) x 128(N); each wave owns a
// 16x128 strip.  Per K-step: A fragment (2x16B contiguous) + all 8 B fragments
// (1x32B contiguous each) are loaded into registers FIRST, then 8 WMMAs run —
// letting the scheduler overlap trailing loads with leading WMMAs.
// Addresses: SGPR base + 32-bit unsigned lane offsets (GVS mode).
// Outputs (any subset, wave-uniform branches):
//   Of : fp32 [M,N], Obn : bf16 [M,N], Obt : bf16 [N,M]
// ---------------------------------------------------------------------------
__global__ void gemm_bt_wmma(const __bf16* __restrict__ Ag, const __bf16* __restrict__ BTg,
                             float* __restrict__ Of, __bf16* __restrict__ Obn,
                             __bf16* __restrict__ Obt,
                             int M, int N, int K, int aBatch, int bBatch, int do_tanh) {
    const int lane = threadIdx.x & 31;
    const int wave = threadIdx.x >> 5;
    const int half = lane >> 4;      // which 16-lane half of the wave
    const int l16  = lane & 15;
    const size_t b = blockIdx.z;

    const __bf16* __restrict__ A  = Ag  + (aBatch ? b * (size_t)M * K : 0);
    const __bf16* __restrict__ BT = BTg + (bBatch ? b * (size_t)N * K : 0);

    const int row0 = blockIdx.y * 128 + wave * 16;
    const int col0 = blockIdx.x * 128;
    if (row0 >= M || col0 >= N) return;

    const v8f zero = {0.f, 0.f, 0.f, 0.f, 0.f, 0.f, 0.f, 0.f};
    v8f acc[8] = {zero, zero, zero, zero, zero, zero, zero, zero};

    // 32-bit lane-relative element offsets (max N*K = 2M elems -> fits u32)
    const unsigned aoff = (unsigned)(row0 + l16) * (unsigned)K + (unsigned)(half * 16);
    unsigned boff[8];
#pragma unroll
    for (int nt = 0; nt < 8; ++nt)
        boff[nt] = (unsigned)(col0 + nt * 16 + l16) * (unsigned)K + (unsigned)(half * 16);

    for (unsigned k0 = 0; k0 < (unsigned)K; k0 += 32) {
        // ---- load phase: A fragment + all 8 B fragments into registers ----
        v8bf a0 = *(const v8bf*)(A + (aoff + k0));
        v8bf a1 = *(const v8bf*)(A + (aoff + k0 + 8));
        v16bf fa = __builtin_shufflevector(a0, a1, 0, 1, 2, 3, 4, 5, 6, 7,
                                                   8, 9, 10, 11, 12, 13, 14, 15);
        v16bf fb[8];
#pragma unroll
        for (int nt = 0; nt < 8; ++nt)
            fb[nt] = *(const v16bf*)(BT + (boff[nt] + k0));

        __builtin_prefetch(A + (aoff + k0 + 64), 0, 1);
        __builtin_prefetch(BT + (boff[0] + k0 + 64), 0, 1);

        // ---- compute phase: 8 WMMAs sharing the A fragment ----
#pragma unroll
        for (int nt = 0; nt < 8; ++nt)
            acc[nt] = __builtin_amdgcn_wmma_f32_16x16x32_bf16(
                false, fa, false, fb[nt], (short)0, acc[nt], false, false);
    }

    // D layout: vgpr r -> M = r + 8*half ; N = l16
    const size_t ob = b * (size_t)M * N;
#pragma unroll
    for (int nt = 0; nt < 8; ++nt) {
        int n = col0 + nt * 16 + l16;
#pragma unroll
        for (int r = 0; r < 8; ++r) {
            int m = row0 + r + half * 8;
            float v = acc[nt][r];
            if (do_tanh) v = tanhf(v);
            if (Of)  Of [ob + (size_t)m * N + n] = v;
            if (Obn) Obn[ob + (size_t)m * N + n] = (__bf16)v;
            if (Obt) Obt[ob + (size_t)n * M + m] = (__bf16)v;
        }
    }
}

// ---------------------------------------------------------------------------
// softmax over last dim (L=256), in place: attn[B*Q, 256]
// ---------------------------------------------------------------------------
__global__ void softmax_rows256(float* __restrict__ attn) {
    __shared__ float red[256];
    int t = threadIdx.x;
    size_t row = blockIdx.x;
    float v = attn[row * 256 + t];
    red[t] = v; __syncthreads();
    for (int s = 128; s > 0; s >>= 1) { if (t < s) red[t] = fmaxf(red[t], red[t + s]); __syncthreads(); }
    float m = red[0]; __syncthreads();
    float e = __expf(v - m);
    red[t] = e; __syncthreads();
    for (int s = 128; s > 0; s >>= 1) { if (t < s) red[t] += red[t + s]; __syncthreads(); }
    float sum = red[0];
    attn[row * 256 + t] = e / sum;
}

// ---------------------------------------------------------------------------
// second softmax: over Q (columns of [Q,L]) for each (b,l); gate with
// combine[b,q,l]; write fp32 attn_map[b,l,q] (d_out) and a bf16 transposed
// copy attnGT[b,q,l] (the BT operand of the final GEMM).
// ---------------------------------------------------------------------------
__global__ void softmax_cols_gate(const float* __restrict__ attn, const float* __restrict__ combine,
                                  float* __restrict__ out_map, __bf16* __restrict__ attnGT) {
    __shared__ float red[256];
    int t = threadIdx.x;
    int b = blockIdx.x / L_;
    int l = blockIdx.x % L_;
    const float* acol = attn    + (size_t)b * Q_ * L_ + l;
    const float* ccol = combine + (size_t)b * Q_ * L_ + l;

    float vals[16];
    float mx = -3.402823466e38f;
#pragma unroll
    for (int j = 0; j < 16; ++j) {
        int q = j * 256 + t;
        float v = acol[(size_t)q * L_];
        vals[j] = v;
        mx = fmaxf(mx, v);
    }
    red[t] = mx; __syncthreads();
    for (int s = 128; s > 0; s >>= 1) { if (t < s) red[t] = fmaxf(red[t], red[t + s]); __syncthreads(); }
    mx = red[0]; __syncthreads();

    float sum = 0.f;
#pragma unroll
    for (int j = 0; j < 16; ++j) { float e = __expf(vals[j] - mx); vals[j] = e; sum += e; }
    red[t] = sum; __syncthreads();
    for (int s = 128; s > 0; s >>= 1) { if (t < s) red[t] += red[t + s]; __syncthreads(); }
    float inv = 1.f / red[0];

    size_t obase = ((size_t)b * L_ + l) * Q_;   // attn_map [B, L, Q]
    size_t tbase = (size_t)b * Q_ * L_ + l;     // attnGT   [B, Q, L]
#pragma unroll
    for (int j = 0; j < 16; ++j) {
        int q = j * 256 + t;
        float g = ccol[(size_t)q * L_];         // combine already tanh'ed in its GEMM
        float r = vals[j] * inv * g;
        out_map[obase + q] = r;
        attnGT[tbase + (size_t)q * L_] = (__bf16)r;
    }
}

// ---------------------------------------------------------------------------
extern "C" void kernel_launch(void* const* d_in, const int* in_sizes, int n_in,
                              void* d_out, int out_size, void* d_ws, size_t ws_size,
                              hipStream_t stream) {
    const float* x         = (const float*)d_in[0];   // [B, IDF, Q]
    const float* context   = (const float*)d_in[1];   // [B, CDF, L]
    const float* w_context = (const float*)d_in[2];   // [IDF, CDF]
    const float* w_img     = (const float*)d_in[3];   // [IDF, IDF]
    const float* w_text    = (const float*)d_in[4];   // [IDF, CDF]

    float* out_wc  = (float*)d_out;                                   // [B, IDF, Q]
    float* out_map = out_wc + (size_t)B_ * IDF_ * Q_;                 // [B, L, Q]

    // ---- workspace carve-up (256B aligned) ----
    char* ws = (char*)d_ws;
    auto carve = [&](size_t bytes) { char* p = ws; ws += (bytes + 255) & ~(size_t)255; return p; };
    __bf16* wt_bf    = (__bf16*)carve((size_t)IDF_ * CDF_ * 2);       // [IDF, CDF]
    __bf16* wc_bf    = (__bf16*)carve((size_t)IDF_ * CDF_ * 2);
    __bf16* wi_bf    = (__bf16*)carve((size_t)IDF_ * IDF_ * 2);
    __bf16* ctxT_bf  = (__bf16*)carve((size_t)B_ * L_ * CDF_ * 2);    // [B, L, CDF]
    __bf16* xT_bf    = (__bf16*)carve((size_t)B_ * Q_ * IDF_ * 2);    // [B, Q, IDF]
    __bf16* textT_bf = (__bf16*)carve((size_t)B_ * L_ * IDF_ * 2);    // [B, L, IDF]
    __bf16* src_bf   = (__bf16*)carve((size_t)B_ * IDF_ * L_ * 2);    // [B, IDF, L]
    __bf16* srcT_bf  = (__bf16*)carve((size_t)B_ * L_ * IDF_ * 2);    // [B, L, IDF]
    __bf16* imgT_bf  = (__bf16*)carve((size_t)B_ * Q_ * IDF_ * 2);    // [B, Q, IDF]
    float*  attn_f   = (float*) carve((size_t)B_ * Q_ * L_ * 4);      // [B, Q, L]
    float*  comb_f   = (float*) carve((size_t)B_ * Q_ * L_ * 4);      // [B, Q, L]
    __bf16* attnGT   = (__bf16*)carve((size_t)B_ * Q_ * L_ * 2);      // [B, Q, L]

    // ---- stage 0: conversions + transposes ----
    cvt_f32_bf16<<<256, 256, 0, stream>>>(w_text,    wt_bf, (size_t)IDF_ * CDF_);
    cvt_f32_bf16<<<256, 256, 0, stream>>>(w_context, wc_bf, (size_t)IDF_ * CDF_);
    cvt_f32_bf16<<<256, 256, 0, stream>>>(w_img,     wi_bf, (size_t)IDF_ * IDF_);
    transpose_f32_to_bf16<<<dim3(L_ / 32, CDF_ / 32, B_), dim3(32, 8), 0, stream>>>(
        context, ctxT_bf, CDF_, L_);
    transpose_f32_to_bf16<<<dim3(Q_ / 32, IDF_ / 32, B_), dim3(32, 8), 0, stream>>>(
        x, xT_bf, IDF_, Q_);

    // ---- stage 1: textT = tanh(w_text @ ctx)^T ; src(+T) = w_context @ ctx ----
    // M=IDF, N=L, K=CDF;  BT = ctxT [L, CDF]
    gemm_bt_wmma<<<dim3(L_ / 128, IDF_ / 128, B_), 256, 0, stream>>>(
        wt_bf, ctxT_bf, nullptr, nullptr, textT_bf, IDF_, L_, CDF_, 0, 1, 1);
    gemm_bt_wmma<<<dim3(L_ / 128, IDF_ / 128, B_), 256, 0, stream>>>(
        wc_bf, ctxT_bf, nullptr, src_bf, srcT_bf, IDF_, L_, CDF_, 0, 1, 0);

    // ---- stage 2: imgT = tanh(w_img @ x)^T  [Q, IDF] ----
    // M=IDF, N=Q, K=IDF;  BT = xT [Q, IDF]
    gemm_bt_wmma<<<dim3(Q_ / 128, IDF_ / 128, B_), 256, 0, stream>>>(
        wi_bf, xT_bf, nullptr, nullptr, imgT_bf, IDF_, Q_, IDF_, 0, 1, 1);

    // ---- stage 3: attn = x^T @ source  [Q, L] fp32 ----
    // M=Q, N=L, K=IDF;  A = xT [Q, IDF], BT = srcT [L, IDF]
    gemm_bt_wmma<<<dim3(L_ / 128, Q_ / 128, B_), 256, 0, stream>>>(
        xT_bf, srcT_bf, attn_f, nullptr, nullptr, Q_, L_, IDF_, 1, 1, 0);

    // ---- stage 4: combine = tanh(img^T @ text)  [Q, L] fp32 ----
    // M=Q, N=L, K=IDF;  A = imgT [Q, IDF], BT = textT [L, IDF]
    gemm_bt_wmma<<<dim3(L_ / 128, Q_ / 128, B_), 256, 0, stream>>>(
        imgT_bf, textT_bf, comb_f, nullptr, nullptr, Q_, L_, IDF_, 1, 1, 1);

    // ---- stage 5: softmax over L (rows), then softmax over Q + gating ----
    softmax_rows256<<<B_ * Q_, 256, 0, stream>>>(attn_f);
    softmax_cols_gate<<<B_ * L_, 256, 0, stream>>>(attn_f, comb_f, out_map, attnGT);

    // ---- stage 6: weighted_context = source @ attnG  [IDF, Q] fp32 -> d_out ----
    // M=IDF, N=Q, K=L;  A = src [IDF, L], BT = attnGT [Q, L]
    gemm_bt_wmma<<<dim3(Q_ / 128, IDF_ / 128, B_), 256, 0, stream>>>(
        src_bf, attnGT, out_wc, nullptr, nullptr, IDF_, Q_, L_, 1, 1, 0);
}
